// RoPEMultiheadAttention_52510270161032
// MI455X (gfx1250) — compile-verified, matrix operand order checked by reading, not run on hardware
//
#include <hip/hip_runtime.h>

// ---------------------------------------------------------------------------
// RoPE multi-head attention for MI455X (gfx1250), wave32 + WMMA bf16 + TDM.
// N=2, L=2048, D=1024, H=16, DH=64.
// ---------------------------------------------------------------------------

#define NB   2
#define LQ   2048
#define DM   1024
#define NH   16
#define DH   64
#define ROWS (NB * LQ)          // 4096 token rows
#define QKVD (3 * DM)           // 3072

typedef __bf16 bf16;
typedef __attribute__((ext_vector_type(16))) __bf16 bf16x16;
typedef __attribute__((ext_vector_type(8)))  float  f32x8;
typedef __attribute__((ext_vector_type(4)))  unsigned int u32x4;
typedef __attribute__((ext_vector_type(8)))  int i32x8;
typedef __attribute__((ext_vector_type(4)))  int i32x4;

__device__ __forceinline__ f32x8 wmma_bf16(bf16x16 a, bf16x16 b, f32x8 c) {
  // D = A(16x32) * B(32x16) + C, f32 accumulate. wave32 only.
  return __builtin_amdgcn_wmma_f32_16x16x32_bf16(false, a, false, b,
                                                 (short)0, c, false, false);
}

__device__ __forceinline__ void wait_tensorcnt0() {
#if __has_builtin(__builtin_amdgcn_s_wait_tensorcnt)
  __builtin_amdgcn_s_wait_tensorcnt(0);
#else
  asm volatile("s_wait_tensorcnt 0x0" ::: "memory");
#endif
}

// ISA 16-bit operand layout: lanes 0-15 own K {k0..k0+7, k0+16..k0+23},
// lanes 16-31 own K {k0+8..k0+15, k0+24..k0+31}: two contiguous 8-elem chunks.
template <typename T>
__device__ __forceinline__ bf16x16 load_frag(const T* __restrict__ src, int ld,
                                             int row, int k0, int lane) {
  bf16x16 f;
  const T* p = src + (size_t)row * ld + k0 + ((lane >= 16) ? 8 : 0);
#pragma unroll
  for (int e = 0; e < 8; ++e) {
    f[e]     = (bf16)p[e];
    f[8 + e] = (bf16)p[16 + e];
  }
  return f;
}

// A-fragment sourced from an LDS f32 row-block (row stride = LQ).
__device__ __forceinline__ bf16x16 load_frag_lds(const float* S, int k0, int lane) {
  bf16x16 f;
  const float* p = S + (lane & 15) * LQ + k0 + ((lane >= 16) ? 8 : 0);
#pragma unroll
  for (int e = 0; e < 8; ++e) {
    f[e]     = (bf16)p[e];
    f[8 + e] = (bf16)p[16 + e];
  }
  return f;
}

// ---------------------------------------------------------------------------
// One-shot f32 -> bf16 cast (x, w_in, w_out pre-conversion). nelem % 4 == 0.
// ---------------------------------------------------------------------------
__global__ __launch_bounds__(256) void cast_bf16(const float* __restrict__ src,
                                                 bf16* __restrict__ dst,
                                                 int nelem) {
  const int i = (blockIdx.x * 256 + threadIdx.x) * 4;
  if (i + 3 < nelem) {
    const float4 v = *(const float4*)(src + i);
    dst[i]     = (bf16)v.x;
    dst[i + 1] = (bf16)v.y;
    dst[i + 2] = (bf16)v.z;
    dst[i + 3] = (bf16)v.w;
  }
}

// ---------------------------------------------------------------------------
// Projection GEMM: C = A @ Bw^T + bias.  A: [M x K] bf16, Bw: [Ncols x K] bf16.
// Fragments load straight from bf16 (pure global_load_b128, no cvt VALU).
// Wave macro-tile 32(M) x 64(N): 2 A-frags + 4 B-frags -> 8 WMMAs per K-step.
// Block = 8 waves -> 32 x 512 output per block.
// ---------------------------------------------------------------------------
__global__ __launch_bounds__(256) void gemm_bias(const bf16* __restrict__ A,
                                                 const bf16* __restrict__ Bw,
                                                 const float* __restrict__ bias,
                                                 float* __restrict__ C,
                                                 int K, int Ncols) {
  const int lane = threadIdx.x & 31;
  const int wave = threadIdx.x >> 5;
  const int m0 = blockIdx.y * 32;
  const int n0 = blockIdx.x * 512 + wave * 64;
  const int ar0 = m0 + (lane & 15);
  const int ar1 = m0 + 16 + (lane & 15);

  f32x8 acc[2][4] = {};
  for (int k0 = 0; k0 < K; k0 += 32) {
    __builtin_prefetch(A + (size_t)ar0 * K + k0 + 512, 0, 0);
    bf16x16 a0 = load_frag(A, K, ar0, k0, lane);
    bf16x16 a1 = load_frag(A, K, ar1, k0, lane);
    bf16x16 b[4];
#pragma unroll
    for (int j = 0; j < 4; ++j)
      b[j] = load_frag(Bw, K, n0 + j * 16 + (lane & 15), k0, lane);
#pragma unroll
    for (int j = 0; j < 4; ++j) {
      acc[0][j] = wmma_bf16(a0, b[j], acc[0][j]);
      acc[1][j] = wmma_bf16(a1, b[j], acc[1][j]);
    }
  }
#pragma unroll
  for (int j = 0; j < 4; ++j) {
    const int col = n0 + j * 16 + (lane & 15);
    const float bv = bias[col];
#pragma unroll
    for (int r = 0; r < 8; ++r) {
      const int m = (lane < 16) ? r : r + 8;
      C[(size_t)(m0 + m) * Ncols + col]      = acc[0][j][r] + bv;
      C[(size_t)(m0 + 16 + m) * Ncols + col] = acc[1][j][r] + bv;
    }
  }
}

// ---------------------------------------------------------------------------
// RoPE on Q,K; pack Q (pre-scaled by DH^-0.5), K as [N,H,L,DH] bf16,
// V transposed as [N,H,DH,L] bf16 so attention B fragments are contiguous.
// ---------------------------------------------------------------------------
__global__ __launch_bounds__(256) void rope_pack(const float* __restrict__ qkv,
                                                 bf16* __restrict__ Qb,
                                                 bf16* __restrict__ Kb,
                                                 bf16* __restrict__ Vt) {
  const int idx = blockIdx.x * 256 + threadIdx.x;       // over N*H*L*DH
  if (idx >= NB * NH * LQ * DH) return;
  const int d = idx & (DH - 1);
  const int l = (idx >> 6) & (LQ - 1);
  const int h = (idx >> 17) & (NH - 1);
  const int n = idx >> 21;

  const size_t rowbase = ((size_t)(n * LQ + l)) * QKVD + h * DH;
  const float qd = qkv[rowbase + d];
  const float kd = qkv[rowbase + DM + d];
  const float vd = qkv[rowbase + 2 * DM + d];

  // theta = pos * 10000^(-(d%32)/32)
  const int i = d & 31;
  const float theta = (float)l * __expf(-(float)i * (9.210340371976184f / 32.0f));
  const float c = __cosf(theta), s = __sinf(theta);
  // rotate_half: d<32 -> -odd[d]; d>=32 -> even[d-32]
  float qp, kp;
  if (d < 32) {
    qp = -qkv[rowbase + 2 * d + 1];
    kp = -qkv[rowbase + DM + 2 * d + 1];
  } else {
    qp = qkv[rowbase + 2 * (d - 32)];
    kp = qkv[rowbase + DM + 2 * (d - 32)];
  }
  const float qr = qd * c + qp * s;
  const float kr = kd * c + kp * s;

  const size_t qk_off = ((size_t)((n * NH + h) * LQ + l)) * DH + d;
  Qb[qk_off] = (bf16)(qr * 0.125f);   // DH^-0.5 folded into Q
  Kb[qk_off] = (bf16)kr;
  Vt[((size_t)((n * NH + h) * DH + d)) * LQ + l] = (bf16)vd;
}

// ---------------------------------------------------------------------------
// TDM: stage one head's 16x64 bf16 Q tile (contiguous rows) into LDS offset 0.
// D#: 2D tensor, data_size=2B, tensor 64 x 2048 (stride 64), tile 64 x 16.
// ---------------------------------------------------------------------------
__device__ __forceinline__ void tdm_load_qtile(const bf16* gQ) {
  const unsigned long long ga = (unsigned long long)(uintptr_t)gQ;
  u32x4 g0;
  g0[0] = 1u;                                        // count=1, user D#
  g0[1] = 0u;                                        // lds_addr = 0 (qtile @ base)
  g0[2] = (unsigned)(ga & 0xFFFFFFFFu);              // global_addr[31:0]
  g0[3] = (unsigned)((ga >> 32) & 0x01FFFFFFu) | 0x80000000u;  // addr[56:32]|type=2
  i32x8 g1;
  g1[0] = 0x00010000;      // workgroup_mask=0, data_size=1 (2 bytes)
  g1[1] = (DH << 16);      // tensor_dim0 = 64 (bits 63:48)
  g1[2] = (LQ << 16);      // tensor_dim0 hi=0, tensor_dim1 = 2048 (bits 95:80)
  g1[3] = (DH << 16);      // tensor_dim1 hi=0, tile_dim0 = 64 (bits 127:112)
  g1[4] = 16;              // tile_dim1 = 16, tile_dim2 = 0
  g1[5] = DH;              // tensor_dim0_stride = 64 (bits 191:160)
  g1[6] = 0;               // stride hi, tensor_dim1_stride lo
  g1[7] = 0;
  i32x4 gz = {0, 0, 0, 0};
#if __has_builtin(__builtin_amdgcn_tensor_load_to_lds)
#if __clang_major__ >= 23
  i32x8 gz8 = {0, 0, 0, 0, 0, 0, 0, 0};
  __builtin_amdgcn_tensor_load_to_lds(g0, g1, gz, gz, gz8, 0);
#else
  __builtin_amdgcn_tensor_load_to_lds(g0, g1, gz, gz, 0);
#endif
  wait_tensorcnt0();
#else
  (void)g0; (void)g1; (void)gz;
#endif
}

// ---------------------------------------------------------------------------
// Attention. One block = (batch n, 16-query block). 8 waves.
// Dynamic LDS: qtile (TDM-staged) + S[16][2048] + AM[16][2048] + reductions
// + P@V partials.  Heads sequential -> plain stores for attn_mean.
// Context written in bf16 so out_proj reads without conversion.
// ---------------------------------------------------------------------------
__global__ __launch_bounds__(256) void attention(const bf16* __restrict__ Qb,
                                                 const bf16* __restrict__ Kb,
                                                 const bf16* __restrict__ Vt,
                                                 bf16* __restrict__ ctx,
                                                 float* __restrict__ amean) {
  extern __shared__ char smem_raw[];
  bf16*  qtile = (bf16*)smem_raw;                    // 16*64 bf16 = 2 KB @ LDS 0
  float* S     = (float*)(smem_raw + 2048);          // 16*2048 scores / probs
  float* AM    = S + 16 * LQ;                        // 16*2048 head-mean accum
  float* red   = AM + 16 * LQ;                       // 256 partials + 16 rows
  float* pvp   = red + 272;                          // 4 * 256 P@V partials

  const int tid  = threadIdx.x;
  const int lane = tid & 31;
  const int wave = tid >> 5;
  const int n  = blockIdx.x >> 7;          // 0..1
  const int q0 = (blockIdx.x & 127) * 16;  // query block start
  const int r  = tid >> 4;                 // softmax: row 0..15
  const int g  = tid & 15;                 // softmax: 16 threads per row

  for (int i = tid; i < 16 * LQ; i += 256) AM[i] = 0.0f;
  __syncthreads();

  for (int h = 0; h < NH; ++h) {
    const bf16* Qh = Qb + ((size_t)(n * NH + h)) * LQ * DH;
    const bf16* Kh = Kb + ((size_t)(n * NH + h)) * LQ * DH;
    const bf16* Vh = Vt + ((size_t)(n * NH + h)) * DH * LQ;

    // --- stage Q tile into LDS via Tensor Data Mover (wave 0 issues) ------
    if (wave == 0) {
#if __has_builtin(__builtin_amdgcn_tensor_load_to_lds)
      tdm_load_qtile(Qh + (size_t)q0 * DH);
#else
      for (int i = lane; i < 16 * DH; i += 32) qtile[i] = Qh[(size_t)q0 * DH + i];
#endif
    }
    __syncthreads();

    // --- S = (Q*scale) @ K^T : each wave owns 256 key columns -------------
    // 2 column tiles per iteration: 4 B-frag loads, then 4 WMMAs.
    bf16x16 a0 = load_frag(qtile, DH, lane & 15, 0, lane);
    bf16x16 a1 = load_frag(qtile, DH, lane & 15, 32, lane);
    for (int t = 0; t < 16; t += 2) {
      const int c0 = wave * 256 + t * 16;
      const int c1 = c0 + 16;
      bf16x16 b00 = load_frag(Kh, DH, c0 + (lane & 15), 0, lane);
      bf16x16 b01 = load_frag(Kh, DH, c0 + (lane & 15), 32, lane);
      bf16x16 b10 = load_frag(Kh, DH, c1 + (lane & 15), 0, lane);
      bf16x16 b11 = load_frag(Kh, DH, c1 + (lane & 15), 32, lane);
      f32x8 s0 = {}, s1 = {};
      s0 = wmma_bf16(a0, b00, s0);
      s1 = wmma_bf16(a0, b10, s1);
      s0 = wmma_bf16(a1, b01, s0);
      s1 = wmma_bf16(a1, b11, s1);
#pragma unroll
      for (int rr = 0; rr < 8; ++rr) {
        const int m = (lane < 16) ? rr : rr + 8;
        S[m * LQ + c0 + (lane & 15)] = s0[rr];
        S[m * LQ + c1 + (lane & 15)] = s1[rr];
      }
    }
    __syncthreads();

    // --- stable softmax over 2048 columns, 16 threads per row -------------
    float mx = -3.4e38f;
    for (int c = g; c < LQ; c += 16) mx = fmaxf(mx, S[r * LQ + c]);
    red[tid] = mx;
    __syncthreads();
    if (g == 0) {
      float m2 = red[tid];
#pragma unroll
      for (int j = 1; j < 16; ++j) m2 = fmaxf(m2, red[tid + j]);
      red[256 + r] = m2;
    }
    __syncthreads();
    const float rowmax = red[256 + r];
    float psum = 0.0f;
    for (int c = g; c < LQ; c += 16) {
      const float e = __expf(S[r * LQ + c] - rowmax);
      S[r * LQ + c] = e;
      psum += e;
    }
    __syncthreads();
    red[tid] = psum;
    __syncthreads();
    if (g == 0) {
      float t2 = red[tid];
#pragma unroll
      for (int j = 1; j < 16; ++j) t2 += red[tid + j];
      red[256 + r] = t2;
    }
    __syncthreads();
    const float inv = 1.0f / red[256 + r];
    for (int c = g; c < LQ; c += 16) {
      const float p = S[r * LQ + c] * inv;
      S[r * LQ + c] = p;
      AM[r * LQ + c] += p * (1.0f / NH);
    }
    __syncthreads();   // probs final before P@V reads them

    // --- out_tile(16x64) = P(16x2048) @ V(2048x64) ------------------------
    // all 8 waves: dh-tile = wave&3, K-half = wave>>2; LDS partial reduce.
    {
      const int dh0  = (wave & 3) * 16;
      const int kbeg = (wave >> 2) * (LQ / 2);
      f32x8 acc = {};
      for (int k0 = kbeg; k0 < kbeg + LQ / 2; k0 += 64) {
        bf16x16 pa0 = load_frag_lds(S, k0, lane);
        bf16x16 pa1 = load_frag_lds(S, k0 + 32, lane);
        bf16x16 pb0 = load_frag(Vh, LQ, dh0 + (lane & 15), k0, lane);
        bf16x16 pb1 = load_frag(Vh, LQ, dh0 + (lane & 15), k0 + 32, lane);
        acc = wmma_bf16(pa0, pb0, acc);
        acc = wmma_bf16(pa1, pb1, acc);
      }
      if (wave >= 4) {
#pragma unroll
        for (int rr = 0; rr < 8; ++rr) {
          const int m = (lane < 16) ? rr : rr + 8;
          pvp[(wave - 4) * 256 + m * 16 + (lane & 15)] = acc[rr];
        }
      }
      __syncthreads();
      if (wave < 4) {
#pragma unroll
        for (int rr = 0; rr < 8; ++rr) {
          const int m = (lane < 16) ? rr : rr + 8;
          const float v = acc[rr] + pvp[wave * 256 + m * 16 + (lane & 15)];
          ctx[((size_t)(n * LQ + q0 + m)) * DM + h * DH + dh0 + (lane & 15)] = (bf16)v;
        }
      }
    }
    __syncthreads();   // S / qtile reused next head
  }

  // --- attn.mean(axis=1) -> second output region --------------------------
  for (int i = tid; i < 16 * LQ; i += 256) {
    const int rr = i >> 11, cc = i & (LQ - 1);
    amean[((size_t)(n * LQ) + q0 + rr) * LQ + cc] = AM[i];
  }
}

// ---------------------------------------------------------------------------
extern "C" void kernel_launch(void* const* d_in, const int* in_sizes, int n_in,
                              void* d_out, int out_size, void* d_ws, size_t ws_size,
                              hipStream_t stream) {
  const float* x     = (const float*)d_in[0];
  const float* w_in  = (const float*)d_in[1];
  const float* b_in  = (const float*)d_in[2];
  const float* w_out = (const float*)d_in[3];
  const float* b_out = (const float*)d_in[4];

  float* out   = (float*)d_out;                         // [N,L,D]
  float* amean = out + (size_t)ROWS * DM;               // [N,L,L]

  // Workspace layout (~101 MB total)
  float* qkv   = (float*)d_ws;                          // 4096*3072 f32
  bf16*  Qb    = (bf16*)(qkv + (size_t)ROWS * QKVD);    // [N,H,L,DH] bf16
  bf16*  Kb    = Qb + (size_t)NB * NH * LQ * DH;
  bf16*  Vt    = Kb + (size_t)NB * NH * LQ * DH;        // [N,H,DH,L] bf16
  bf16*  xb    = Vt + (size_t)NB * NH * LQ * DH;        // 4096*1024 bf16
  bf16*  w_inb = xb + (size_t)ROWS * DM;                // 3072*1024 bf16
  bf16*  w_outb= w_inb + (size_t)QKVD * DM;             // 1024*1024 bf16
  bf16*  ctxb  = w_outb + (size_t)DM * DM;              // 4096*1024 bf16

  // 0) one-shot bf16 pre-conversion of x and both weight matrices
  cast_bf16<<<(ROWS * DM) / 1024, 256, 0, stream>>>(x, xb, ROWS * DM);
  cast_bf16<<<(QKVD * DM) / 1024, 256, 0, stream>>>(w_in, w_inb, QKVD * DM);
  cast_bf16<<<(DM * DM) / 1024, 256, 0, stream>>>(w_out, w_outb, DM * DM);

  // 1) QKV projection: 4096x3072, blocks of 32x512
  gemm_bias<<<dim3(QKVD / 512, ROWS / 32), 256, 0, stream>>>(xb, w_inb, b_in, qkv,
                                                             DM, QKVD);
  // 2) RoPE + bf16 packing (V transposed)
  rope_pack<<<(NB * NH * LQ * DH) / 256, 256, 0, stream>>>(qkv, Qb, Kb, Vt);
  // 3) attention + head-mean probs (~269 KB dynamic LDS per workgroup)
  const size_t smem_bytes = 2048 + (size_t)(32 * LQ + 272 + 1024) * sizeof(float);
  attention<<<NB * (LQ / 16), 256, smem_bytes, stream>>>(Qb, Kb, Vt, ctxb, amean);
  // 4) output projection: 4096x1024
  gemm_bias<<<dim3(DM / 512, ROWS / 32), 256, 0, stream>>>(ctxb, w_outb, b_out, out,
                                                           DM, DM);
}
